// MultiHeadAttention_52072183497002
// MI455X (gfx1250) — matrix operand order repaired and checked by reading, not compiled, not run on hardware
//
#include <hip/hip_runtime.h>
#include <hip/hip_bf16.h>

// ---------------------------------------------------------------------------
// MHA forward for MI455X (gfx1250, wave32):
//   qkv = x @ W_qkv^T + b_qkv        (bf16 WMMA GEMM, async-LDS double buffer)
//   flash attention (causal)         (bf16 WMMA, TDM K-tile, 64-key blocks)
//   out = attn @ W_out^T + b_out     (bf16 WMMA GEMM, f32 out)
// ---------------------------------------------------------------------------

typedef __attribute__((ext_vector_type(16))) __bf16 v16bf;
typedef __attribute__((ext_vector_type(8)))  __bf16 v8bf;
typedef __attribute__((ext_vector_type(8)))  float  v8f;
typedef __attribute__((ext_vector_type(4)))  unsigned int u32x4;
typedef __attribute__((ext_vector_type(8)))  int i32x8;
typedef __attribute__((ext_vector_type(4)))  int i32x4;

#define D_MODEL 2048
#define N_HEADS 16
#define D_K     128
#define QKV_LD  (3 * D_MODEL)   // 6144

#if defined(__has_builtin)
#if __has_builtin(__builtin_amdgcn_tensor_load_to_lds)
#define HAVE_TDM 1
#endif
#endif

__device__ __forceinline__ v16bf mk16(const __bf16* lo, const __bf16* hi) {
    union { v16bf v; v8bf h[2]; } u;
    u.h[0] = *(const v8bf*)lo;
    u.h[1] = *(const v8bf*)hi;
    return u.v;
}

__device__ __forceinline__ v8f wmma_bf16(v16bf a, v16bf b, v8f c) {
    // D = A(16x32 bf16) * B(32x16 bf16) + C(16x16 f32)
    return __builtin_amdgcn_wmma_f32_16x16x32_bf16(
        /*neg_a=*/false, a, /*neg_b=*/false, b,
        /*c_mod=*/(short)0, c, /*reuse_a=*/false, /*reuse_b=*/false);
}

__device__ __forceinline__ v8f zero8() {
    v8f v;
    #pragma unroll
    for (int i = 0; i < 8; ++i) v[i] = 0.0f;
    return v;
}

// Async global -> LDS copy of 16 bytes per lane (ASYNCcnt-tracked).
__device__ __forceinline__ void async_b128(const __bf16* g, __bf16* l) {
    unsigned lds = (unsigned)(unsigned long long)l;      // low 32 bits = LDS offset
    unsigned long long ga = (unsigned long long)g;
    asm volatile("global_load_async_to_lds_b128 %0, %1, off"
                 :: "v"(lds), "v"(ga) : "memory");
}

__device__ __forceinline__ void wait_async0() {
    asm volatile("s_wait_asynccnt 0" ::: "memory");
}

// ---------------------------------------------------------------------------
// fp32 -> bf16 conversion (grid-stride)
// ---------------------------------------------------------------------------
__global__ void f32_to_bf16_kernel(const float* __restrict__ in,
                                   __bf16* __restrict__ out, long n) {
    long i = (long)blockIdx.x * blockDim.x + threadIdx.x;
    long stride = (long)gridDim.x * blockDim.x;
    for (; i < n; i += stride) out[i] = (__bf16)in[i];
}

// ---------------------------------------------------------------------------
// GEMM: C[M,N] = A[M,K](bf16,rm) @ B[N,K]^T(bf16,rm) + bias[N]
// Block tile 128x128, BK=32, 256 threads = 8 waves in 4(M) x 2(N).
// Each wave: 32x64 tile = 2 A-frags x 4 B-frags -> 8 WMMA per K-step.
// A/B tiles stream via async global->LDS with 2-deep double buffering.
// ---------------------------------------------------------------------------
template <typename OutT>
__global__ __launch_bounds__(256)
void gemm_bf16_wmma_kernel(const __bf16* __restrict__ A,
                           const __bf16* __restrict__ B,
                           const float* __restrict__ bias,
                           OutT* __restrict__ C,
                           int M, int N, int K) {
    __shared__ __bf16 As[2][128 * 32];   // [m][k]
    __shared__ __bf16 Bs[2][128 * 32];   // [n][k]

    const int tid  = threadIdx.x;
    const int lane = tid & 31;
    const int wave = tid >> 5;
    const int wm   = wave & 3;        // 32-row slice of 128
    const int wn   = wave >> 2;       // 64-col slice of 128
    const int l16  = lane & 15;
    const int hi   = (lane >= 16);

    const int m0 = blockIdx.y * 128;
    const int n0 = blockIdx.x * 128;

    // stage one 128x32 A tile + one 128x32 B tile: 512 chunks of 16B each
    auto stage = [&](int kt, int buf) {
        #pragma unroll
        for (int i = 0; i < 2; ++i) {
            int chunk = i * 256 + tid;            // 0..511
            int row = chunk >> 2, seg = chunk & 3;
            async_b128(&A[(size_t)(m0 + row) * K + kt + seg * 8],
                       &As[buf][row * 32 + seg * 8]);
            async_b128(&B[(size_t)(n0 + row) * K + kt + seg * 8],
                       &Bs[buf][row * 32 + seg * 8]);
        }
    };

    v8f acc[2][4];
    #pragma unroll
    for (int i = 0; i < 2; ++i)
        #pragma unroll
        for (int j = 0; j < 4; ++j) acc[i][j] = zero8();

    stage(0, 0);
    wait_async0();
    __syncthreads();

    const int nk = K / 32;
    for (int t = 0; t < nk; ++t) {
        const int cur = t & 1;
        if (t + 1 < nk) stage((t + 1) * 32, cur ^ 1);   // prefetch next tiles

        const __bf16* as = As[cur];
        const __bf16* bs = Bs[cur];

        // A fragments: lane<16 -> k {0..7,16..23}; lane>=16 -> {8..15,24..31}
        const int ak0 = hi ? 8 : 0;
        v16bf af[2];
        #pragma unroll
        for (int i = 0; i < 2; ++i) {
            int arow = wm * 32 + i * 16 + l16;
            af[i] = mk16(&as[arow * 32 + ak0], &as[arow * 32 + ak0 + 16]);
        }
        // B fragments: col = l16; lane<16 -> k {0..7,16..23}; lane>=16 -> {8..15,24..31}
        const int bk0 = hi ? 8 : 0;
        #pragma unroll
        for (int j = 0; j < 4; ++j) {
            int bcol = wn * 64 + j * 16 + l16;
            v16bf bfrag = mk16(&bs[bcol * 32 + bk0], &bs[bcol * 32 + bk0 + 16]);
            #pragma unroll
            for (int i = 0; i < 2; ++i)
                acc[i][j] = wmma_bf16(af[i], bfrag, acc[i][j]);
        }

        wait_async0();          // next tiles landed in LDS
        __syncthreads();        // everyone done reading current tiles
    }

    // epilogue: C/D layout row = r + (hi?8:0), col = l16
    #pragma unroll
    for (int i = 0; i < 2; ++i)
        #pragma unroll
        for (int j = 0; j < 4; ++j) {
            int col = n0 + wn * 64 + j * 16 + l16;
            float bv = bias[col];
            #pragma unroll
            for (int r = 0; r < 8; ++r) {
                int row = m0 + wm * 32 + i * 16 + r + (hi ? 8 : 0);
                C[(size_t)row * N + col] = (OutT)(acc[i][j][r] + bv);
            }
        }
}

// ---------------------------------------------------------------------------
// Flash attention (causal). qkv: [B*T, 6144] bf16 (q|k|v per row).
// Grid: (T/64, B*H). Block: 128 threads = 4 waves; wave owns 16 query rows.
// 64-key blocks: K tile (64x128, row stride 6144) loaded by the Tensor Data
// Mover when available; V tile staged transposed concurrently. 32 WMMAs and
// one softmax/rescale pass per 64 keys.
// ---------------------------------------------------------------------------
__global__ __launch_bounds__(128)
void flash_attn_kernel(const __bf16* __restrict__ qkv,
                       __bf16* __restrict__ out, int T) {
    __shared__ __bf16 Ks[64 * 128];     // [key][d]            16 KB
    __shared__ __bf16 Vt[128 * 64];     // [d][key] transposed 16 KB
    __shared__ __bf16 Ps[4][16 * 64];   // per-wave P buffer    8 KB

    const int tid  = threadIdx.x;
    const int lane = tid & 31;
    const int wave = tid >> 5;
    const int l16  = lane & 15;
    const int hi   = (lane >= 16);

    const int b  = blockIdx.y >> 4;
    const int h  = blockIdx.y & 15;
    const int q0 = blockIdx.x * 64;
    const int qRowBase = q0 + wave * 16;

    const size_t rowBase = (size_t)b * T;
    const int qOff = h * D_K;
    const int kOff = D_MODEL + h * D_K;
    const int vOff = 2 * D_MODEL + h * D_K;

    // ---- load Q A-fragments (16x128 per wave, kept in registers)
    v16bf qfrag[4];
    {
        const size_t qrow = rowBase + qRowBase + l16;
        const __bf16* qp = qkv + qrow * QKV_LD + qOff;
        #pragma unroll
        for (int c = 0; c < 4; ++c) {
            int d0 = c * 32 + (hi ? 8 : 0);
            qfrag[c] = mk16(qp + d0, qp + d0 + 16);
        }
    }

    float m[8], l[8];
    v8f o[8];
    #pragma unroll
    for (int r = 0; r < 8; ++r) { m[r] = -3.0e38f; l[r] = 0.0f; }
    #pragma unroll
    for (int t = 0; t < 8; ++t) o[t] = zero8();

    const float scale = 0.08838834764831845f;  // 1/sqrt(128)
    const int nkb = blockIdx.x + 1;            // causal: 64-key blocks < q0+64

    for (int kb = 0; kb < nkb; ++kb) {
        const int key0 = kb * 64;

        // ---- stage K tile [64 keys][128 d] via TDM (fallback: async copies)
#ifdef HAVE_TDM
        if (wave == 0) {
            unsigned ldsA = (unsigned)(unsigned long long)&Ks[0];
            unsigned long long ga =
                (unsigned long long)(qkv + (rowBase + key0) * QKV_LD + kOff);
            u32x4 g0;
            g0[0] = 1u;                                    // count=1 (valid D#)
            g0[1] = ldsA;                                  // lds_addr
            g0[2] = (unsigned)ga;                          // global_addr[31:0]
            g0[3] = ((unsigned)(ga >> 32) & 0x01FFFFFFu)   // global_addr[56:32]
                    | 0x80000000u;                         // type=2 ("image")
            i32x8 g1;
            g1[0] = (int)(1u << 16);          // data_size=1 (2 bytes/elem)
            g1[1] = (int)(128u << 16);        // tensor_dim0 = 128
            g1[2] = (int)(64u << 16);         // tensor_dim1 = 64
            g1[3] = (int)(128u << 16);        // tile_dim0 = 128
            g1[4] = 64;                       // tile_dim1 = 64, tile_dim2 = 0
            g1[5] = QKV_LD;                   // tensor_dim0_stride = 6144
            g1[6] = 0;
            g1[7] = 0;
            i32x4 g2 = {};
            i32x4 g3 = {};
            i32x8 g4 = {};                    // extra group (6-arg toolchain form)
            __builtin_amdgcn_tensor_load_to_lds(g0, g1, g2, g3, g4, 0);
            __builtin_amdgcn_s_wait_tensorcnt(0);
        }
#else
        #pragma unroll
        for (int i = 0; i < 8; ++i) {
            int chunk = i * 128 + tid;               // 1024 x 16B chunks
            int kr = chunk >> 4, seg = chunk & 15;
            async_b128(&qkv[(rowBase + key0 + kr) * QKV_LD + kOff + seg * 8],
                       &Ks[kr * 128 + seg * 8]);
        }
        wait_async0();
#endif
        // ---- stage V transposed: Vt[d][key]
        #pragma unroll
        for (int i = 0; i < 8; ++i) {
            int chunk = i * 128 + tid;               // 1024 x 16B chunks
            int kr = chunk >> 4, seg = chunk & 15;
            uint4 raw = *(const uint4*)&qkv[(rowBase + key0 + kr) * QKV_LD + vOff + seg * 8];
            union { uint4 u; __bf16 e[8]; } cv; cv.u = raw;
            #pragma unroll
            for (int e = 0; e < 8; ++e)
                Vt[(seg * 8 + e) * 64 + kr] = cv.e[e];
        }
        __syncthreads();

        // ---- S = Q K^T : 4 N-tiles x 4 K-chunks (16 WMMA)
        v8f s[4];
        #pragma unroll
        for (int j = 0; j < 4; ++j) s[j] = zero8();
        #pragma unroll
        for (int j = 0; j < 4; ++j) {
            int kcol = j * 16 + l16;               // key column
            #pragma unroll
            for (int c = 0; c < 4; ++c) {
                int d0 = c * 32 + (hi ? 8 : 0);    // B-frag K offset (table order)
                v16bf kf = mk16(&Ks[kcol * 128 + d0], &Ks[kcol * 128 + d0 + 16]);
                s[j] = wmma_bf16(qfrag[c], kf, s[j]);
            }
        }

        // ---- scale + causal mask
        #pragma unroll
        for (int j = 0; j < 4; ++j) {
            int col = key0 + j * 16 + l16;
            #pragma unroll
            for (int r = 0; r < 8; ++r) {
                int row = qRowBase + r + (hi ? 8 : 0);
                float v = s[j][r] * scale;
                s[j][r] = (col > row) ? -3.0e38f : v;
            }
        }

        // ---- online softmax (row reduction across 16-lane groups)
        #pragma unroll
        for (int r = 0; r < 8; ++r) {
            float rmax = fmaxf(fmaxf(s[0][r], s[1][r]), fmaxf(s[2][r], s[3][r]));
            #pragma unroll
            for (int off = 8; off > 0; off >>= 1)
                rmax = fmaxf(rmax, __shfl_xor(rmax, off, 16));
            float mnew  = fmaxf(m[r], rmax);
            float alpha = __expf(m[r] - mnew);
            float rsum = 0.0f;
            #pragma unroll
            for (int j = 0; j < 4; ++j) {
                float p = __expf(s[j][r] - mnew);
                s[j][r] = p;
                rsum += p;
            }
            #pragma unroll
            for (int off = 8; off > 0; off >>= 1)
                rsum += __shfl_xor(rsum, off, 16);
            l[r] = l[r] * alpha + rsum;
            m[r] = mnew;
            #pragma unroll
            for (int t = 0; t < 8; ++t) o[t][r] *= alpha;
        }

        // ---- C-layout -> A-layout for P via per-wave LDS buffer
        __bf16* P = &Ps[wave][0];                  // [16 rows][64 keys]
        #pragma unroll
        for (int j = 0; j < 4; ++j)
            #pragma unroll
            for (int r = 0; r < 8; ++r)
                P[(r + (hi ? 8 : 0)) * 64 + j * 16 + l16] = (__bf16)s[j][r];
        asm volatile("s_wait_dscnt 0" ::: "memory");

        const int pk0 = hi ? 8 : 0;
        v16bf pfrag[2];
        #pragma unroll
        for (int kk = 0; kk < 2; ++kk)
            pfrag[kk] = mk16(&P[l16 * 64 + kk * 32 + pk0],
                             &P[l16 * 64 + kk * 32 + pk0 + 16]);

        // ---- O += P V : 8 d-tiles x 2 key-chunks (16 WMMA)
        const int vk0 = hi ? 8 : 0;
        #pragma unroll
        for (int t = 0; t < 8; ++t) {
            int d = t * 16 + l16;
            #pragma unroll
            for (int kk = 0; kk < 2; ++kk) {
                v16bf vf = mk16(&Vt[d * 64 + kk * 32 + vk0],
                                &Vt[d * 64 + kk * 32 + vk0 + 16]);
                o[t] = wmma_bf16(pfrag[kk], vf, o[t]);
            }
        }
        __syncthreads();
    }

    // ---- normalize + store (bf16, [B,T,C] layout)
    #pragma unroll
    for (int t = 0; t < 8; ++t) {
        int d = t * 16 + l16;
        #pragma unroll
        for (int r = 0; r < 8; ++r) {
            int row = qRowBase + r + (hi ? 8 : 0);
            out[(rowBase + row) * D_MODEL + h * D_K + d] = (__bf16)(o[t][r] / l[r]);
        }
    }
}

// ---------------------------------------------------------------------------
// Launcher
// ---------------------------------------------------------------------------
extern "C" void kernel_launch(void* const* d_in, const int* in_sizes, int n_in,
                              void* d_out, int out_size, void* d_ws, size_t ws_size,
                              hipStream_t stream) {
    const float* x     = (const float*)d_in[0];  // [2,2048,2048]
    const float* W_qkv = (const float*)d_in[1];  // [6144,2048]
    const float* b_qkv = (const float*)d_in[2];  // [6144]
    const float* W_out = (const float*)d_in[3];  // [2048,2048]
    const float* b_out = (const float*)d_in[4];  // [2048]
    float* out = (float*)d_out;                  // [2,2048,2048] f32

    const int Bsz = 2, T = 2048;
    const size_t M = (size_t)Bsz * T;            // 4096 rows

    // workspace layout (bf16)
    __bf16* xb    = (__bf16*)d_ws;                         // M x 2048
    __bf16* wqkvb = xb    + M * D_MODEL;                   // 6144 x 2048
    __bf16* woutb = wqkvb + (size_t)QKV_LD * D_MODEL;      // 2048 x 2048
    __bf16* qkv   = woutb + (size_t)D_MODEL * D_MODEL;     // M x 6144
    __bf16* attn  = qkv   + M * QKV_LD;                    // M x 2048

    // 1) convert inputs to bf16
    f32_to_bf16_kernel<<<2048, 256, 0, stream>>>(x,     xb,    (long)(M * D_MODEL));
    f32_to_bf16_kernel<<<2048, 256, 0, stream>>>(W_qkv, wqkvb, (long)QKV_LD * D_MODEL);
    f32_to_bf16_kernel<<<2048, 256, 0, stream>>>(W_out, woutb, (long)D_MODEL * D_MODEL);

    // 2) qkv = x @ W_qkv^T + b_qkv   (M=4096, N=6144, K=2048)
    gemm_bf16_wmma_kernel<__bf16>
        <<<dim3(QKV_LD / 128, M / 128), 256, 0, stream>>>(
            xb, wqkvb, b_qkv, qkv, (int)M, QKV_LD, D_MODEL);

    // 3) flash attention (causal), per (64 q-rows, head)
    flash_attn_kernel<<<dim3(T / 64, Bsz * N_HEADS), 128, 0, stream>>>(qkv, attn, T);

    // 4) out = attn @ W_out^T + b_out   (M=4096, N=2048, K=2048), f32 output
    gemm_bf16_wmma_kernel<float>
        <<<dim3(D_MODEL / 128, M / 128), 256, 0, stream>>>(
            attn, woutb, b_out, out, (int)M, D_MODEL, D_MODEL);
}
